// StereoBase_14516989461272
// MI455X (gfx1250) — compile-verified, matrix operand order checked by reading, not run on hardware
//
#include <hip/hip_runtime.h>
#include <hip/hip_bf16.h>

typedef __attribute__((ext_vector_type(16))) __bf16 v16bf;
typedef __attribute__((ext_vector_type(8)))  float  v8f;

// ---------------- problem constants ----------------
constexpr int C_IN = 96;
constexpr int H_   = 80;
constexpr int W_   = 184;
constexpr int HWn  = H_ * W_;        // 14720
constexpr int D_   = 48;             // MAXD/4
constexpr int CV_  = 72;             // 8 + 2*32
constexpr int HID_ = 128;
constexpr int GDIM_  = 1314;         // 2*(72*9+9)
constexpr int GDIMP_ = 1344;         // padded to 42*32
constexpr int KCONV  = 2304;         // 256*9
constexpr int NT_  = HWn / 16;       // 920
constexpr int HO_  = H_ * 4;
constexpr int WO_  = W_ * 4;

// =====================================================================
// 1) cl = w_concat @ match_left ; crf = w_concat @ match_right  (32x96)
// =====================================================================
__global__ __launch_bounds__(256)
void k_concat_proj(const float* __restrict__ ml, const float* __restrict__ mr,
                   const float* __restrict__ wcat,
                   float* __restrict__ cl, float* __restrict__ crf)
{
    int idx = blockIdx.x * blockDim.x + threadIdx.x;
    if (idx >= 32 * HWn) return;
    int oc = idx / HWn, n = idx - oc * HWn;
    float a = 0.f, b = 0.f;
    for (int c = 0; c < C_IN; ++c) {
        float w = wcat[oc * C_IN + c];
        a += w * ml[(size_t)c * HWn + n];
        b += w * mr[(size_t)c * HWn + n];
    }
    cl[idx] = a;
    crf[idx] = b;
}

// =====================================================================
// 2) geo volume in (h,w,cv,d) layout: [gwc(8) | left(32) | right(32)]
// =====================================================================
__global__ __launch_bounds__(64)
void k_build_geo(const float* __restrict__ ml, const float* __restrict__ mr,
                 const float* __restrict__ cl, const float* __restrict__ crf,
                 float* __restrict__ geo)
{
    __shared__ float lft[C_IN];
    __shared__ float cls[32];
    int w = blockIdx.x, h = blockIdx.y, t = threadIdx.x;
    int base = h * W_ + w;
    if (t < C_IN) lft[t] = ml[(size_t)t * HWn + base];
    if (t < 32)  cls[t] = cl[(size_t)t * HWn + base];
    __syncthreads();
    int d = t;
    if (d >= D_) return;
    float* g = geo + (size_t)base * CV_ * D_;
    int ws = w - d;
    if (ws < 0) {
        for (int cv = 0; cv < CV_; ++cv) g[cv * D_ + d] = 0.f;
        return;
    }
    int rb = h * W_ + ws;
    for (int gi = 0; gi < 8; ++gi) {
        float acc = 0.f;
        for (int c = 0; c < 12; ++c) {
            int ch = gi * 12 + c;
            acc += lft[ch] * mr[(size_t)ch * HWn + rb];
        }
        g[gi * D_ + d] = acc * (1.f / 12.f);
    }
    for (int cc = 0; cc < 32; ++cc) g[(8 + cc) * D_ + d] = cls[cc];
    for (int cc = 0; cc < 32; ++cc) g[(40 + cc) * D_ + d] = crf[(size_t)cc * HWn + rb];
}

// =====================================================================
// 3) classifier 3D conv (72ch,3x3x3) + softmax over d + expectation
// =====================================================================
__global__ __launch_bounds__(64)
void k_classifier(const float* __restrict__ geo, const float* __restrict__ wcls,
                  float* __restrict__ initd, float* __restrict__ disp)
{
    __shared__ float wl[CV_ * 27];
    __shared__ float lg[D_];
    int w = blockIdx.x, h = blockIdx.y, t = threadIdx.x;
    for (int i = t; i < CV_ * 27; i += 64) wl[i] = wcls[i];
    __syncthreads();
    if (t < D_) {
        float acc = 0.f;
        for (int kh = 0; kh < 3; ++kh) {
            int hh = h + kh - 1; if (hh < 0 || hh >= H_) continue;
            for (int kw = 0; kw < 3; ++kw) {
                int ww2 = w + kw - 1; if (ww2 < 0 || ww2 >= W_) continue;
                const float* gc = geo + (size_t)(hh * W_ + ww2) * CV_ * D_;
                for (int ic = 0; ic < CV_; ++ic) {
                    const float* vp = gc + ic * D_;
                    const float* wp = wl + ic * 27 + kh * 3 + kw;   // +kd*9
                    if (t - 1 >= 0) acc += vp[t - 1] * wp[0];
                    acc += vp[t] * wp[9];
                    if (t + 1 < D_) acc += vp[t + 1] * wp[18];
                }
            }
        }
        lg[t] = acc;
    }
    __syncthreads();
    if (t == 0) {
        float m = -1e30f;
        for (int i = 0; i < D_; ++i) m = fmaxf(m, lg[i]);
        float s = 0.f, e = 0.f;
        for (int i = 0; i < D_; ++i) {
            float ex = __expf(lg[i] - m);
            s += ex; e += ex * (float)i;
        }
        float id = e / s;
        int n = h * W_ + w;
        initd[n] = id;
        disp[n]  = id;
    }
}

// =====================================================================
// 4) corr[h,w,v] = (1/sqrt(96)) * sum_c L[c,h,w]*R[c,h,v]
// =====================================================================
__global__ __launch_bounds__(256)
void k_corr(const float* __restrict__ ml, const float* __restrict__ mr,
            float* __restrict__ corr)
{
    int idx = blockIdx.x * blockDim.x + threadIdx.x;
    if (idx >= HWn * W_) return;
    int v = idx % W_;
    int hw = idx / W_;
    int h = hw / W_;
    float acc = 0.f;
    for (int c = 0; c < C_IN; ++c)
        acc += ml[(size_t)c * HWn + hw] * mr[(size_t)c * HWn + h * W_ + v];
    corr[idx] = acc * 0.102062072616f;   // 1/sqrt(96)
}

// =====================================================================
// elementwise helpers
// =====================================================================
__global__ __launch_bounds__(256)
void k_tanh(const float* __restrict__ in, float* __restrict__ out, int n)
{
    int i = blockIdx.x * blockDim.x + threadIdx.x;
    if (i < n) out[i] = tanhf(in[i]);
}
__global__ __launch_bounds__(256)
void k_dispenc(const float* __restrict__ wde, const float* __restrict__ disp,
               float* __restrict__ df)
{
    int idx = blockIdx.x * blockDim.x + threadIdx.x;
    if (idx >= 64 * HWn) return;
    int c = idx / HWn, n = idx - c * HWn;
    df[idx] = fmaxf(wde[c] * disp[n], 0.f);
}

// =====================================================================
// 5) lookup: gfeat[(k,n)] (K padded with zero rows 1314..1343)
//    order: geo0(648) corr0(9) geo1(648) corr1(9)
// =====================================================================
__device__ __forceinline__ float samp_lin(const float* vol, int L, float x)
{
    float x0 = floorf(x);
    int i0 = (int)x0;
    float w1 = x - x0;
    float t0 = (i0 >= 0 && i0 < L) ? vol[i0] : 0.f;
    int i1 = i0 + 1;
    float t1 = (i1 >= 0 && i1 < L) ? vol[i1] : 0.f;
    return t0 * (1.f - w1) + t1 * w1;
}
__device__ __forceinline__ float samp_lin_pool(const float* vol, int L, float x)
{
    float x0 = floorf(x);
    int i0 = (int)x0;
    float w1 = x - x0;
    float t0 = (i0 >= 0 && i0 < L) ? 0.5f * (vol[2 * i0] + vol[2 * i0 + 1]) : 0.f;
    int i1 = i0 + 1;
    float t1 = (i1 >= 0 && i1 < L) ? 0.5f * (vol[2 * i1] + vol[2 * i1 + 1]) : 0.f;
    return t0 * (1.f - w1) + t1 * w1;
}
__global__ __launch_bounds__(256)
void k_lookup(const float* __restrict__ geo, const float* __restrict__ corr,
              const float* __restrict__ disp, float* __restrict__ gfeat)
{
    int hw = blockIdx.x * blockDim.x + threadIdx.x;
    if (hw >= HWn) return;
    int w = hw % W_;
    float d2 = disp[hw];
    const float* gcol = geo + (size_t)hw * CV_ * D_;
    const float* ccol = corr + (size_t)hw * W_;
    for (int cv = 0; cv < CV_; ++cv) {
        const float* vol = gcol + cv * D_;
        for (int k = 0; k < 9; ++k) {
            float x = d2 + (float)(k - 4);
            gfeat[(size_t)(cv * 9 + k) * HWn + hw] = samp_lin(vol, D_, x);
        }
    }
    for (int k = 0; k < 9; ++k) {
        float x = ((float)w - d2) + (float)(k - 4);
        gfeat[(size_t)(648 + k) * HWn + hw] = samp_lin(ccol, W_, x);
    }
    for (int cv = 0; cv < CV_; ++cv) {
        const float* vol = gcol + cv * D_;
        for (int k = 0; k < 9; ++k) {
            float x = d2 * 0.5f + (float)(k - 4);
            gfeat[(size_t)(657 + cv * 9 + k) * HWn + hw] = samp_lin_pool(vol, D_ / 2, x);
        }
    }
    for (int k = 0; k < 9; ++k) {
        float x = ((float)w - d2) * 0.5f + (float)(k - 4);
        gfeat[(size_t)(1305 + k) * HWn + hw] = samp_lin_pool(ccol, W_ / 2, x);
    }
    for (int k = GDIM_; k < GDIMP_; ++k)            // zero pad rows
        gfeat[(size_t)k * HWn + hw] = 0.f;
}

// =====================================================================
// A-matrix prepack into WMMA fragment order (bf16)
//   mode 0: dense (zero-pad k >= Korig); mode 1: conv reorder k = r*256+ic
// =====================================================================
__global__ __launch_bounds__(256)
void k_pack_a(const float* __restrict__ A, __bf16* __restrict__ out,
              int MT, int KSTEPS, int Korig, int mode)
{
    int idx = blockIdx.x * blockDim.x + threadIdx.x;
    int total = MT * KSTEPS * 512;
    if (idx >= total) return;
    int j = idx & 15;
    int lane = (idx >> 4) & 31;
    int frag = idx >> 9;
    int ks = frag % KSTEPS;
    int mtile = frag / KSTEPS;
    int hl = lane >> 4;
    int m = mtile * 16 + (lane & 15);
    int kk = (j & 7) + 8 * (2 * (j >> 3) + hl);
    int k = ks * 32 + kk;
    float v = 0.f;
    if (mode == 1) {
        int r = k >> 8, ic = k & 255;
        v = A[(size_t)m * KCONV + ic * 9 + r];
    } else if (k < Korig) {
        v = A[(size_t)m * Korig + k];
    }
    out[idx] = (__bf16)v;
}

__device__ __forceinline__ float actf(float v, int ACT)
{
    if (ACT == 1) return fmaxf(v, 0.f);
    if (ACT == 2) return 1.f / (1.f + __expf(-v));
    if (ACT == 3) return tanhf(v);
    return v;
}
__device__ __forceinline__ float sigf(float v) { return 1.f / (1.f + __expf(-v)); }

// =====================================================================
// dense/split-B GEMM:  C(MT*16 x HW) = act(A * B(KxHW))
// =====================================================================
template <int MT, int KSTEPS, int ACT, bool SPLIT>
__global__ __launch_bounds__(128)
void k_gemm_dense(const __bf16* __restrict__ Apk,
                  const float* __restrict__ B0,
                  const float* __restrict__ B1, int K0,
                  float* __restrict__ C)
{
    int lane = threadIdx.x;
    int hl = lane >> 4, mlane = lane & 15;
    int ntile = blockIdx.x * blockDim.y + threadIdx.y;
    int n = ntile * 16 + mlane;
    const v16bf* Af = (const v16bf*)Apk;

    v8f acc[MT] = {};
    for (int ks = 0; ks < KSTEPS; ++ks) {
        int k0 = ks * 32;
        const float* Bp;
        if (SPLIT) Bp = (k0 < K0) ? (B0 + (size_t)k0 * HWn) : (B1 + (size_t)(k0 - K0) * HWn);
        else       Bp = B0 + (size_t)k0 * HWn;
        const float* bbase = Bp + n + (size_t)(8 * hl) * HWn;
        v16bf bfrag;
#pragma unroll
        for (int j = 0; j < 8; ++j) bfrag[j]     = (__bf16)bbase[(size_t)j * HWn];
#pragma unroll
        for (int j = 0; j < 8; ++j) bfrag[8 + j] = (__bf16)bbase[(size_t)(16 + j) * HWn];
#pragma unroll
        for (int mt = 0; mt < MT; ++mt) {
            v16bf afrag = Af[(size_t)(mt * KSTEPS + ks) * 32 + lane];
            acc[mt] = __builtin_amdgcn_wmma_f32_16x16x32_bf16(
                false, afrag, false, bfrag, (short)0, acc[mt], false, false);
        }
    }
#pragma unroll
    for (int mt = 0; mt < MT; ++mt) {
#pragma unroll
        for (int r = 0; r < 8; ++r) {
            int m = mt * 16 + r + 8 * hl;
            C[(size_t)m * HWn + n] = actf(acc[mt][r], ACT);
        }
    }
}

// =====================================================================
// fused z+r conv GEMM (shared im2col B): M=128 each, K=2304 reordered.
//   z    = sigmoid(conv([net;motion], wz) + cz)
//   rnet = sigmoid(conv([net;motion], wr) + cr) * net
// =====================================================================
__global__ __launch_bounds__(128)
void k_gemm_conv3x3_zr(const __bf16* __restrict__ ApkZ,
                       const __bf16* __restrict__ ApkR,
                       const float* __restrict__ P0,   // net
                       const float* __restrict__ P1,   // motion
                       const float* __restrict__ AddZ, // cz
                       const float* __restrict__ AddR, // cr
                       float* __restrict__ Zout,
                       float* __restrict__ RNout)
{
    int lane = threadIdx.x;
    int hl = lane >> 4, mlane = lane & 15;
    int ntile = blockIdx.x * blockDim.y + threadIdx.y;
    int n = ntile * 16 + mlane;
    int h = n / W_, w = n - h * W_;
    const v16bf* AfZ = (const v16bf*)ApkZ;
    const v16bf* AfR = (const v16bf*)ApkR;
    constexpr int KSTEPS = 72;

    v8f accz[8] = {};
    v8f accr[8] = {};
    for (int r = 0; r < 9; ++r) {
        int kh = r / 3 - 1, kw = r - (r / 3) * 3 - 1;
        int hh = h + kh, ww2 = w + kw;
        float msk = (hh >= 0 && hh < H_ && ww2 >= 0 && ww2 < W_) ? 1.f : 0.f;
        int hc = min(max(hh, 0), H_ - 1);
        int wc = min(max(ww2, 0), W_ - 1);
        int pos = hc * W_ + wc;
#pragma unroll
        for (int s = 0; s < 8; ++s) {                      // ic0 = 32*s
            int ic0 = s * 32;
            const float* src = (ic0 < 128) ? (P0 + (size_t)ic0 * HWn)
                                           : (P1 + (size_t)(ic0 - 128) * HWn);
            const float* bbase = src + pos + (size_t)(8 * hl) * HWn;
            v16bf bfrag;
#pragma unroll
            for (int j = 0; j < 8; ++j) bfrag[j]     = (__bf16)(bbase[(size_t)j * HWn] * msk);
#pragma unroll
            for (int j = 0; j < 8; ++j) bfrag[8 + j] = (__bf16)(bbase[(size_t)(16 + j) * HWn] * msk);
            int ks = r * 8 + s;
#pragma unroll
            for (int mt = 0; mt < 8; ++mt) {
                v16bf az = AfZ[(size_t)(mt * KSTEPS + ks) * 32 + lane];
                accz[mt] = __builtin_amdgcn_wmma_f32_16x16x32_bf16(
                    false, az, false, bfrag, (short)0, accz[mt], false, false);
                v16bf ar = AfR[(size_t)(mt * KSTEPS + ks) * 32 + lane];
                accr[mt] = __builtin_amdgcn_wmma_f32_16x16x32_bf16(
                    false, ar, false, bfrag, (short)0, accr[mt], false, false);
            }
        }
    }
#pragma unroll
    for (int mt = 0; mt < 8; ++mt) {
#pragma unroll
        for (int r = 0; r < 8; ++r) {
            int m = mt * 16 + r + 8 * hl;
            size_t o = (size_t)m * HWn + n;
            Zout[o]  = sigf(accz[mt][r] + AddZ[o]);
            RNout[o] = sigf(accr[mt][r] + AddR[o]) * P0[o];
        }
    }
}

// =====================================================================
// q conv GEMM + fused GRU update:
//   q = tanh(conv([rnet;motion], wq) + cq);  net = (1-z)*net + z*q
// (B operand never reads `net`, so the in-place net update is safe)
// =====================================================================
__global__ __launch_bounds__(128)
void k_gemm_conv3x3_qgru(const __bf16* __restrict__ Apk,
                         const float* __restrict__ P0,   // rnet
                         const float* __restrict__ P1,   // motion
                         const float* __restrict__ Add,  // cq
                         const float* __restrict__ Zbuf, // z
                         float* __restrict__ net)
{
    int lane = threadIdx.x;
    int hl = lane >> 4, mlane = lane & 15;
    int ntile = blockIdx.x * blockDim.y + threadIdx.y;
    int n = ntile * 16 + mlane;
    int h = n / W_, w = n - h * W_;
    const v16bf* Af = (const v16bf*)Apk;
    constexpr int KSTEPS = 72;

    v8f acc[8] = {};
    for (int r = 0; r < 9; ++r) {
        int kh = r / 3 - 1, kw = r - (r / 3) * 3 - 1;
        int hh = h + kh, ww2 = w + kw;
        float msk = (hh >= 0 && hh < H_ && ww2 >= 0 && ww2 < W_) ? 1.f : 0.f;
        int hc = min(max(hh, 0), H_ - 1);
        int wc = min(max(ww2, 0), W_ - 1);
        int pos = hc * W_ + wc;
#pragma unroll
        for (int s = 0; s < 8; ++s) {
            int ic0 = s * 32;
            const float* src = (ic0 < 128) ? (P0 + (size_t)ic0 * HWn)
                                           : (P1 + (size_t)(ic0 - 128) * HWn);
            const float* bbase = src + pos + (size_t)(8 * hl) * HWn;
            v16bf bfrag;
#pragma unroll
            for (int j = 0; j < 8; ++j) bfrag[j]     = (__bf16)(bbase[(size_t)j * HWn] * msk);
#pragma unroll
            for (int j = 0; j < 8; ++j) bfrag[8 + j] = (__bf16)(bbase[(size_t)(16 + j) * HWn] * msk);
            int ks = r * 8 + s;
#pragma unroll
            for (int mt = 0; mt < 8; ++mt) {
                v16bf afrag = Af[(size_t)(mt * KSTEPS + ks) * 32 + lane];
                acc[mt] = __builtin_amdgcn_wmma_f32_16x16x32_bf16(
                    false, afrag, false, bfrag, (short)0, acc[mt], false, false);
            }
        }
    }
#pragma unroll
    for (int mt = 0; mt < 8; ++mt) {
#pragma unroll
        for (int r = 0; r < 8; ++r) {
            int m = mt * 16 + r + 8 * hl;
            size_t o = (size_t)m * HWn + n;
            float qv = tanhf(acc[mt][r] + Add[o]);
            float zv = Zbuf[o];
            net[o] = (1.f - zv) * net[o] + zv * qv;
        }
    }
}

// =====================================================================
// 6) delta: disp += conv3x3(net, w_delta)  (M=1, VALU)
// =====================================================================
__global__ __launch_bounds__(256)
void k_delta(const float* __restrict__ net, const float* __restrict__ wdel,
             float* __restrict__ disp)
{
    int n = blockIdx.x * blockDim.x + threadIdx.x;
    if (n >= HWn) return;
    int h = n / W_, w = n - h * W_;
    float acc = 0.f;
    for (int ic = 0; ic < HID_; ++ic) {
        const float* src = net + (size_t)ic * HWn;
        const float* wk = wdel + ic * 9;
        for (int kh = 0; kh < 3; ++kh) {
            int hh = h + kh - 1; if (hh < 0 || hh >= H_) continue;
            for (int kw = 0; kw < 3; ++kw) {
                int ww2 = w + kw - 1; if (ww2 < 0 || ww2 >= W_) continue;
                acc += src[hh * W_ + ww2] * wk[kh * 3 + kw];
            }
        }
    }
    disp[n] += acc;
}

// =====================================================================
// 7) fused spx conv-transpose + softmax(9) + context_upsample
// =====================================================================
__global__ __launch_bounds__(256)
void k_upsample(const float* __restrict__ mf, const float* __restrict__ wspx,
                const float* __restrict__ disp, float* __restrict__ out,
                int reluMask)
{
    int idx = blockIdx.x * blockDim.x + threadIdx.x;
    if (idx >= HO_ * WO_) return;
    int y = idx / WO_, x = idx - y * WO_;
    int h0 = y >> 2, w0 = x >> 2;
    int ky = 3 - (y & 3), kx = 3 - (x & 3);
    float logits[9];
    float mx = -1e30f;
    for (int k = 0; k < 9; ++k) {
        float acc = 0.f;
        for (int ic = 0; ic < 32; ++ic) {
            float m = mf[(size_t)ic * HWn + h0 * W_ + w0];
            if (reluMask) m = fmaxf(m, 0.f);
            acc += m * wspx[((k * 32 + ic) * 4 + ky) * 4 + kx];
        }
        logits[k] = acc;
        mx = fmaxf(mx, acc);
    }
    float s = 0.f;
    for (int k = 0; k < 9; ++k) { logits[k] = __expf(logits[k] - mx); s += logits[k]; }
    float inv = 1.f / s;
    float o = 0.f;
    for (int i = 0; i < 3; ++i) {
        int hh = h0 + i - 1;
        for (int j = 0; j < 3; ++j) {
            int ww2 = w0 + j - 1;
            float dv = (hh >= 0 && hh < H_ && ww2 >= 0 && ww2 < W_)
                           ? disp[hh * W_ + ww2] * 4.f : 0.f;
            o += dv * logits[i * 3 + j] * inv;
        }
    }
    out[idx] = o;
}

// =====================================================================
// host launch
// =====================================================================
extern "C" void kernel_launch(void* const* d_in, const int* in_sizes, int n_in,
                              void* d_out, int out_size, void* d_ws, size_t ws_size,
                              hipStream_t stream)
{
    (void)in_sizes; (void)n_in; (void)out_size; (void)ws_size;
    const float* match_left  = (const float*)d_in[0];
    const float* match_right = (const float*)d_in[1];
    const float* w_concat    = (const float*)d_in[2];
    const float* w_cls       = (const float*)d_in[3];
    const float* net0        = (const float*)d_in[4];
    const float* cz          = (const float*)d_in[5];
    const float* cq          = (const float*)d_in[6];
    const float* cr          = (const float*)d_in[7];
    const float* w_geo       = (const float*)d_in[8];
    const float* w_dispenc   = (const float*)d_in[9];
    const float* w_mix       = (const float*)d_in[10];
    const float* wz          = (const float*)d_in[11];
    const float* wr          = (const float*)d_in[12];
    const float* wq          = (const float*)d_in[13];
    const float* w_delta     = (const float*)d_in[14];
    const float* w_mask      = (const float*)d_in[15];
    const float* w_spx       = (const float*)d_in[16];
    const float* w_spx0      = (const float*)d_in[17];

    float* ws = (float*)d_ws;
    size_t off = 0;
    float* geo    = ws + off; off += (size_t)HWn * CV_ * D_;   // 50,872,320
    float* corr   = ws + off; off += (size_t)HWn * W_;         //  2,708,480
    float* cl     = ws + off; off += (size_t)32 * HWn;
    float* crf    = ws + off; off += (size_t)32 * HWn;
    float* initd  = ws + off; off += HWn;
    float* disp   = ws + off; off += HWn;
    float* net    = ws + off; off += (size_t)HID_ * HWn;
    float* gfeat  = ws + off; off += (size_t)GDIMP_ * HWn;     // 19,783,680
    float* gf     = ws + off; off += (size_t)HID_ * HWn;
    float* df     = ws + off; off += (size_t)64 * HWn;
    float* motion = ws + off; off += (size_t)HID_ * HWn;
    float* zb     = ws + off; off += (size_t)HID_ * HWn;
    float* rnet   = ws + off; off += (size_t)HID_ * HWn;
    float* maskf  = ws + off; off += (size_t)32 * HWn;

    // packed bf16 weight region (32B-aligned: off*4 is a multiple of 32)
    __bf16* pkbase = (__bf16*)(ws + off);
    size_t poff = 0;
    __bf16* apk_geo  = pkbase + poff; poff += (size_t)8 * 42 * 512;  // 172,032
    __bf16* apk_mix  = pkbase + poff; poff += (size_t)8 * 6 * 512;
    __bf16* apk_mask = pkbase + poff; poff += (size_t)2 * 4 * 512;
    __bf16* apk_z    = pkbase + poff; poff += (size_t)8 * 72 * 512;  // 294,912
    __bf16* apk_r    = pkbase + poff; poff += (size_t)8 * 72 * 512;
    __bf16* apk_q    = pkbase + poff; poff += (size_t)8 * 72 * 512;

    // ---------- weight prepack (deterministic, every call) ----------
    k_pack_a<<<(8 * 42 * 512) / 256, 256, 0, stream>>>(w_geo,  apk_geo,  8, 42, GDIM_, 0);
    k_pack_a<<<(8 * 6  * 512) / 256, 256, 0, stream>>>(w_mix,  apk_mix,  8, 6,  192,   0);
    k_pack_a<<<(2 * 4  * 512) / 256, 256, 0, stream>>>(w_mask, apk_mask, 2, 4,  128,   0);
    k_pack_a<<<(8 * 72 * 512) / 256, 256, 0, stream>>>(wz,     apk_z,    8, 72, KCONV, 1);
    k_pack_a<<<(8 * 72 * 512) / 256, 256, 0, stream>>>(wr,     apk_r,    8, 72, KCONV, 1);
    k_pack_a<<<(8 * 72 * 512) / 256, 256, 0, stream>>>(wq,     apk_q,    8, 72, KCONV, 1);

    // ---------- stage A: volumes ----------
    k_concat_proj<<<1840, 256, 0, stream>>>(match_left, match_right, w_concat, cl, crf);
    k_build_geo<<<dim3(W_, H_), 64, 0, stream>>>(match_left, match_right, cl, crf, geo);
    k_classifier<<<dim3(W_, H_), 64, 0, stream>>>(geo, w_cls, initd, disp);
    k_corr<<<(HWn * W_) / 256, 256, 0, stream>>>(match_left, match_right, corr);
    k_tanh<<<(HID_ * HWn) / 256, 256, 0, stream>>>(net0, net, HID_ * HWn);

    // ---------- stage B: 4 GRU iterations ----------
    dim3 gblk(32, 4);               // 4 waves, each owns one 16-wide N tile
    const int ggrd = NT_ / 4;       // 230 blocks cover N = 14720
    for (int it = 0; it < 4; ++it) {
        k_lookup<<<58, 256, 0, stream>>>(geo, corr, disp, gfeat);
        // gf = relu(w_geo @ gfeat)            M=128 K=1344(pad)
        k_gemm_dense<8, 42, 1, false><<<ggrd, gblk, 0, stream>>>(
            apk_geo, gfeat, nullptr, 0, gf);
        k_dispenc<<<(64 * HWn) / 256, 256, 0, stream>>>(w_dispenc, disp, df);
        // motion = relu(w_mix @ [gf; df])     M=128 K=192 (split 128/64)
        k_gemm_dense<8, 6, 1, true><<<ggrd, gblk, 0, stream>>>(
            apk_mix, gf, df, 128, motion);
        // fused: z = sig(convz+cz); rnet = sig(convr+cr)*net  (shared B)
        k_gemm_conv3x3_zr<<<ggrd, gblk, 0, stream>>>(
            apk_z, apk_r, net, motion, cz, cr, zb, rnet);
        // fused: q = tanh(convq+cq); net = (1-z)*net + z*q
        k_gemm_conv3x3_qgru<<<ggrd, gblk, 0, stream>>>(
            apk_q, rnet, motion, cq, zb, net);
        k_delta<<<58, 256, 0, stream>>>(net, w_delta, disp);
    }
    // mask_feat = relu(w_mask @ net)   M=32 K=128
    k_gemm_dense<2, 4, 1, false><<<ggrd, gblk, 0, stream>>>(
        apk_mask, net, nullptr, 0, maskf);

    // ---------- stage C: upsampled outputs ----------
    float* out = (float*)d_out;
    k_upsample<<<(HO_ * WO_) / 256, 256, 0, stream>>>(cl, w_spx0, initd, out, 1);
    k_upsample<<<(HO_ * WO_) / 256, 256, 0, stream>>>(maskf, w_spx, disp, out + HO_ * WO_, 0);
}
